// Attention_88785563943008
// MI455X (gfx1250) — compile-verified, matrix operand order checked by reading, not run on hardware
//
#include <hip/hip_runtime.h>
#include <hip/hip_bf16.h>

// ---------------------------------------------------------------------------
// MI455X attention layer: RMSNorm -> QKV GEMM -> RoPE -> flash attention ->
// output projection (+residual, +mask). GEMMs on v_wmma_f32_16x16x32_bf16 with
// async global->LDS staging (ASYNCcnt) and double buffering.
// T = D_MODEL = 2048, N_H = 16, D_H = 128.
// ---------------------------------------------------------------------------

typedef __attribute__((ext_vector_type(8)))  __bf16 bf16x8;
typedef __attribute__((ext_vector_type(16))) __bf16 bf16x16;
typedef __attribute__((ext_vector_type(8)))  float  f32x8;

#define TT 2048
#define DM 2048
#define NH 16
#define DH 128

// Low 32 bits of a generic pointer to LDS == wave-relative LDS byte address.
__device__ __forceinline__ unsigned lds_addr(const void* p) {
  return (unsigned)(size_t)p;
}

// CDNA5 async copy: 16 bytes per lane, global -> LDS, tracked by ASYNCcnt.
__device__ __forceinline__ void async_copy_b128(unsigned lds_off, const void* g) {
  unsigned long long ga = (unsigned long long)(size_t)g;
  asm volatile("global_load_async_to_lds_b128 %0, %1, off"
               :: "v"(lds_off), "v"(ga) : "memory");
}

// Load a 16x32 bf16 fragment in the CDNA5 WMMA A/B register layout from a
// row-major matrix (16 rows starting at `row`, 32 contiguous K at `col`).
// Per-lane: lane&15 = row; VGPR0-3 <- K[(lane>=16?8:0)+0..7], VGPR4-7 <- K[16+...].
__device__ __forceinline__ bf16x16 load_frag(const __bf16* base, int ld,
                                             int row, int col, int lane) {
  const __bf16* p = base + (size_t)(row + (lane & 15)) * ld + col + ((lane >> 4) << 3);
  bf16x8 lo = *(const bf16x8*)(p);
  bf16x8 hi = *(const bf16x8*)(p + 16);
  bf16x16 f;
#pragma unroll
  for (int i = 0; i < 8; ++i) { f[i] = lo[i]; f[i + 8] = hi[i]; }
  return f;
}

__device__ __forceinline__ void wmma_bf16(f32x8& acc, bf16x16 a, bf16x16 b) {
  acc = __builtin_amdgcn_wmma_f32_16x16x32_bf16(false, a, false, b,
                                                (short)0, acc, false, false);
}

// ---------------------------------------------------------------------------
// RMSNorm: h = ln_w * x * rsqrt(mean(x^2)+eps), output bf16. One block per row.
// ---------------------------------------------------------------------------
__global__ void __launch_bounds__(256) rmsnorm_bf16_kernel(
    const float* __restrict__ x, const float* __restrict__ w,
    __bf16* __restrict__ h, int D) {
  __shared__ float red[8];
  int row = blockIdx.x;
  const float* xr = x + (size_t)row * D;
  float s = 0.f;
  for (int i = threadIdx.x; i < D; i += 256) { float v = xr[i]; s += v * v; }
  s += __shfl_xor(s, 1);  s += __shfl_xor(s, 2);
  s += __shfl_xor(s, 4);  s += __shfl_xor(s, 8);  s += __shfl_xor(s, 16);
  int lane = threadIdx.x & 31, wv = threadIdx.x >> 5;
  if (lane == 0) red[wv] = s;
  __syncthreads();
  if (wv == 0) {
    float t = (lane < 8) ? red[lane] : 0.f;
    t += __shfl_xor(t, 1); t += __shfl_xor(t, 2); t += __shfl_xor(t, 4);
    if (lane == 0) red[0] = t;
  }
  __syncthreads();
  float inv = rsqrtf(red[0] / (float)D + 1e-5f);
  for (int i = threadIdx.x; i < D; i += 256)
    h[(size_t)row * D + i] = (__bf16)(xr[i] * w[i] * inv);
}

// fp32 -> bf16 elementwise
__global__ void __launch_bounds__(256) f2bf_kernel(
    const float* __restrict__ src, __bf16* __restrict__ dst, int n) {
  int i = blockIdx.x * 256 + threadIdx.x;
  if (i < n) dst[i] = (__bf16)src[i];
}

// ---------------------------------------------------------------------------
// Tiled WMMA GEMM: C[M][N](f32) = A[M][K](bf16) @ B[N][K]^T(bf16)
// Block tile 64x256, 8 waves (2x4), 32x64 per wave. Double-buffered LDS fed by
// GLOBAL_LOAD_ASYNC_TO_LDS_B128 (5 async instrs per wave per K-slab).
// Optional epilogue: C = residual + (mask[row] ? acc : 0).
// ---------------------------------------------------------------------------
#define BM 64
#define BN 256
#define BK 32
#define LDSK (BK + 8)

__global__ void __launch_bounds__(256) gemm_bf16_wmma_kernel(
    const __bf16* __restrict__ A, const __bf16* __restrict__ B,
    float* __restrict__ C, int M, int N, int K,
    const float* __restrict__ residual, const int* __restrict__ mask) {
  __shared__ __bf16 As[2][BM][LDSK];
  __shared__ __bf16 Bs[2][BN][LDSK];
  int tid = threadIdx.x;
  int lane = tid & 31, wave = tid >> 5;
  int wm = wave >> 2, wn = wave & 3;      // waves: 2 (M) x 4 (N)
  int bm = blockIdx.y * BM, bn = blockIdx.x * BN;

  int ra = tid >> 2, ca = (tid & 3) * 8;  // A tile: 64 rows x 4 chunks

  auto issue_tile = [&](int buf, int k0) {
    async_copy_b128(lds_addr(&As[buf][ra][ca]),
                    A + (size_t)(bm + ra) * K + k0 + ca);
#pragma unroll
    for (int it = 0; it < 4; ++it) {      // B tile: 256 rows x 4 chunks
      int idx = tid + it * 256;
      int rb = idx >> 2, cb = (idx & 3) * 8;
      async_copy_b128(lds_addr(&Bs[buf][rb][cb]),
                      B + (size_t)(bn + rb) * K + k0 + cb);
    }
  };

  f32x8 acc[2][4] = {};
  int nk = K / BK;
  issue_tile(0, 0);
  for (int kt = 0; kt < nk; ++kt) {
    int cur = kt & 1;
    if (kt + 1 < nk) {
      issue_tile(1 - cur, (kt + 1) * BK);
      asm volatile("s_wait_asynccnt 0x5" ::: "memory");  // prev slab landed
    } else {
      asm volatile("s_wait_asynccnt 0x0" ::: "memory");
    }
    __syncthreads();

    bf16x16 af[2], bfm[4];
#pragma unroll
    for (int i = 0; i < 2; ++i)
      af[i] = load_frag(&As[cur][0][0], LDSK, wm * 32 + i * 16, 0, lane);
#pragma unroll
    for (int j = 0; j < 4; ++j)
      bfm[j] = load_frag(&Bs[cur][0][0], LDSK, wn * 64 + j * 16, 0, lane);
#pragma unroll
    for (int i = 0; i < 2; ++i)
#pragma unroll
      for (int j = 0; j < 4; ++j) wmma_bf16(acc[i][j], af[i], bfm[j]);
    __syncthreads();   // all waves done reading before buffer is re-filled
  }

  // C/D layout: VGPR e, lane L -> (m = e + 8*(L>=16), n = L&15)
  int mofs = (lane >= 16) ? 8 : 0, nofs = lane & 15;
#pragma unroll
  for (int i = 0; i < 2; ++i)
#pragma unroll
    for (int j = 0; j < 4; ++j)
#pragma unroll
      for (int e = 0; e < 8; ++e) {
        int r = bm + wm * 32 + i * 16 + e + mofs;
        int c = bn + wn * 64 + j * 16 + nofs;
        float v = acc[i][j][e];
        if (residual) v = residual[(size_t)r * N + c] + (mask[r] ? v : 0.f);
        C[(size_t)r * N + c] = v;
      }
}

// ---------------------------------------------------------------------------
// RoPE + repack to per-head bf16 [h][t][dh]
// ---------------------------------------------------------------------------
__global__ void __launch_bounds__(256) rope_repack_kernel(
    const float* __restrict__ src, const float* __restrict__ cosT,
    const float* __restrict__ sinT, __bf16* __restrict__ dst, int T) {
  size_t idx = (size_t)blockIdx.x * 256 + threadIdx.x;   // over T*DM
  int d = (int)(idx & (DH - 1));
  int h = (int)((idx >> 7) & (NH - 1));
  int t = (int)(idx >> 11);
  float v = src[idx];
  float other = (d < DH / 2) ? -src[idx + DH / 2] : src[idx - DH / 2];
  float c = cosT[(size_t)t * DH + d], s = sinT[(size_t)t * DH + d];
  dst[((size_t)h * T + t) * DH + d] = (__bf16)(v * c + other * s);
}

// V repack transposed: Vt[h][d][t] (bf16) from V[t][h*DH+d] (f32)
__global__ void __launch_bounds__(256) v_transpose_kernel(
    const float* __restrict__ src, __bf16* __restrict__ dst, int T) {
  size_t idx = (size_t)blockIdx.x * 256 + threadIdx.x;
  int d = (int)(idx & (DH - 1));
  int h = (int)((idx >> 7) & (NH - 1));
  int t = (int)(idx >> 11);
  dst[((size_t)h * DH + d) * T + t] = (__bf16)src[idx];
}

// ---------------------------------------------------------------------------
// Flash attention: 1 wave per (head, 16-query tile). 32-key tiles:
// QK^T via 8 WMMAs, online softmax (half-wave shfl reductions), P transposed
// to A-fragment layout through per-wave LDS, PV via 8 WMMAs.
// ---------------------------------------------------------------------------
__global__ void __launch_bounds__(256) flash_attn_kernel(
    const __bf16* __restrict__ Q, const __bf16* __restrict__ Kmat,
    const __bf16* __restrict__ Vt, const int* __restrict__ am,
    __bf16* __restrict__ O, int T) {
  __shared__ __bf16 pbuf[8][16 * 32];
  int lane = threadIdx.x & 31, wave = threadIdx.x >> 5;
  int job = blockIdx.x * 8 + wave;        // NH * (T/16) jobs
  int nqt = T >> 4;
  int h = job / nqt, qt = job % nqt;
  if (h >= NH) return;

  const __bf16* Qh = Q    + (size_t)h * T * DH;
  const __bf16* Kh = Kmat + (size_t)h * T * DH;
  const __bf16* Vh = Vt   + (size_t)h * DH * T;

  bf16x16 qf[4];
#pragma unroll
  for (int kk = 0; kk < 4; ++kk) qf[kk] = load_frag(Qh, DH, qt * 16, kk * 32, lane);

  f32x8 o[8] = {};
  float mrow[8], lrow[8];
#pragma unroll
  for (int i = 0; i < 8; ++i) { mrow[i] = -3e38f; lrow[i] = 0.f; }

  const float scale = 0.08838834764831845f;   // 1/sqrt(128)
  int mofs = (lane >= 16) ? 8 : 0;
  int nkt = ((qt * 16 + 15) >> 5) + 1;        // causal: key tiles of 32

  for (int kt = 0; kt < nkt; ++kt) {
    int kbase = kt * 32;
    f32x8 s0 = {}, s1 = {};
#pragma unroll
    for (int kk = 0; kk < 4; ++kk) {
      bf16x16 k0 = load_frag(Kh, DH, kbase,      kk * 32, lane);
      bf16x16 k1 = load_frag(Kh, DH, kbase + 16, kk * 32, lane);
      wmma_bf16(s0, qf[kk], k0);
      wmma_bf16(s1, qf[kk], k1);
    }
    int kcol0 = kbase + (lane & 15), kcol1 = kcol0 + 16;
    int am0 = am[kcol0], am1 = am[kcol1];
    __bf16* pb = &pbuf[wave][0];
#pragma unroll
    for (int i = 0; i < 8; ++i) {
      int qrow = qt * 16 + i + mofs;
      float v0 = s0[i] * scale, v1 = s1[i] * scale;
      if (kcol0 > qrow || !am0) v0 = -3e38f;
      if (kcol1 > qrow || !am1) v1 = -3e38f;
      float mx = fmaxf(v0, v1);
      mx = fmaxf(mx, __shfl_xor(mx, 1)); mx = fmaxf(mx, __shfl_xor(mx, 2));
      mx = fmaxf(mx, __shfl_xor(mx, 4)); mx = fmaxf(mx, __shfl_xor(mx, 8));
      float mn = fmaxf(mrow[i], mx);
      float alpha = __expf(mrow[i] - mn);
      mrow[i] = mn;
      float p0 = __expf(v0 - mn), p1 = __expf(v1 - mn);
      float rs = p0 + p1;
      rs += __shfl_xor(rs, 1); rs += __shfl_xor(rs, 2);
      rs += __shfl_xor(rs, 4); rs += __shfl_xor(rs, 8);
      lrow[i] = lrow[i] * alpha + rs;
#pragma unroll
      for (int j = 0; j < 8; ++j) o[j][i] *= alpha;
      int ql = i + mofs;                       // stage P (C-layout -> LDS)
      pb[ql * 32 +      (lane & 15)] = (__bf16)p0;
      pb[ql * 32 + 16 + (lane & 15)] = (__bf16)p1;
    }
    asm volatile("s_wait_dscnt 0x0" ::: "memory");
    bf16x16 pf = load_frag(pb, 32, 0, 0, lane);  // P as 16x32 A-fragment
#pragma unroll
    for (int j = 0; j < 8; ++j) {
      bf16x16 vf = load_frag(Vh, T, j * 16, kbase, lane);   // V^T as B-fragment
      wmma_bf16(o[j], pf, vf);
    }
  }

#pragma unroll
  for (int i = 0; i < 8; ++i) {
    float inv = 1.f / lrow[i];
    int qrow = qt * 16 + i + mofs;
#pragma unroll
    for (int j = 0; j < 8; ++j)
      O[(size_t)qrow * DM + h * DH + j * 16 + (lane & 15)] = (__bf16)(o[j][i] * inv);
  }
}

// ---------------------------------------------------------------------------
extern "C" void kernel_launch(void* const* d_in, const int* in_sizes, int n_in,
                              void* d_out, int out_size, void* d_ws, size_t ws_size,
                              hipStream_t stream) {
  const float* x    = (const float*)d_in[0];
  const float* cosT = (const float*)d_in[1];
  const float* sinT = (const float*)d_in[2];
  const int*   am   = (const int*)  d_in[3];
  const float* ln_w = (const float*)d_in[4];
  const float* Wq   = (const float*)d_in[5];
  const float* Wk   = (const float*)d_in[6];
  const float* Wv   = (const float*)d_in[7];
  const float* Wo   = (const float*)d_in[8];
  float* out = (float*)d_out;

  char* ws = (char*)d_ws;
  const size_t MB = 1024u * 1024u;
  __bf16* h_bf   = (__bf16*)(ws);              // 8 MB  (T*DM bf16)
  __bf16* wq_bf  = (__bf16*)(ws +   8 * MB);
  __bf16* wk_bf  = (__bf16*)(ws +  16 * MB);
  __bf16* wv_bf  = (__bf16*)(ws +  24 * MB);
  __bf16* wo_bf  = (__bf16*)(ws +  32 * MB);
  float*  Qf     = (float*) (ws +  40 * MB);   // 16 MB each
  float*  Kf     = (float*) (ws +  56 * MB);
  float*  Vf     = (float*) (ws +  72 * MB);
  __bf16* Qr     = (__bf16*)(ws +  88 * MB);   // 8 MB each
  __bf16* Kr     = (__bf16*)(ws +  96 * MB);
  __bf16* Vtb    = (__bf16*)(ws + 104 * MB);
  __bf16* attn   = (__bf16*)(ws + 112 * MB);

  rmsnorm_bf16_kernel<<<TT, 256, 0, stream>>>(x, ln_w, h_bf, DM);

  int nW = DM * DM;
  f2bf_kernel<<<nW / 256, 256, 0, stream>>>(Wq, wq_bf, nW);
  f2bf_kernel<<<nW / 256, 256, 0, stream>>>(Wk, wk_bf, nW);
  f2bf_kernel<<<nW / 256, 256, 0, stream>>>(Wv, wv_bf, nW);
  f2bf_kernel<<<nW / 256, 256, 0, stream>>>(Wo, wo_bf, nW);

  dim3 ggrid(DM / BN, TT / BM);
  gemm_bf16_wmma_kernel<<<ggrid, 256, 0, stream>>>(h_bf, wq_bf, Qf, TT, DM, DM, nullptr, nullptr);
  gemm_bf16_wmma_kernel<<<ggrid, 256, 0, stream>>>(h_bf, wk_bf, Kf, TT, DM, DM, nullptr, nullptr);
  gemm_bf16_wmma_kernel<<<ggrid, 256, 0, stream>>>(h_bf, wv_bf, Vf, TT, DM, DM, nullptr, nullptr);

  int nE = TT * DM;
  rope_repack_kernel<<<nE / 256, 256, 0, stream>>>(Qf, cosT, sinT, Qr, TT);
  rope_repack_kernel<<<nE / 256, 256, 0, stream>>>(Kf, cosT, sinT, Kr, TT);
  v_transpose_kernel<<<nE / 256, 256, 0, stream>>>(Vf, Vtb, TT);

  flash_attn_kernel<<<(NH * (TT / 16)) / 8, 256, 0, stream>>>(Qr, Kr, Vtb, am, attn, TT);

  gemm_bf16_wmma_kernel<<<ggrid, 256, 0, stream>>>(attn, wo_bf, out, TT, DM, DM, x, am);
}
